// EncoderLayer_6493990552253
// MI455X (gfx1250) — matrix operand order repaired and checked by reading, not compiled, not run on hardware
//
#include <hip/hip_runtime.h>

typedef __bf16 v16bf __attribute__((ext_vector_type(16)));
typedef float  v8f   __attribute__((ext_vector_type(8)));

#define BQ_STRIDE 20   // LDS B-tile column stride in uints (16 payload + 4 pad; 80B keeps 16B align, spreads banks)

// ---------------------------------------------------------------------------
// fp32 -> bf16 conversion (activations stay row-major)
// ---------------------------------------------------------------------------
__global__ void f32_to_bf16_kernel(const float* __restrict__ in, __bf16* __restrict__ out, int n) {
    int i = blockIdx.x * blockDim.x + threadIdx.x;
    if (i < n) out[i] = (__bf16)in[i];
}

// ---------------------------------------------------------------------------
// Weight pack: W[K,N] fp32 row-major  ->  Wp uint32[N][K/2]
// Wp[n*(K/2)+q] = bf16(W[2q,n]) | bf16(W[2q+1,n])<<16
// This is precisely the WMMA B-operand pair layout, so GEMM-side B fragment
// loads become contiguous ds_load_b128's.
// ---------------------------------------------------------------------------
__global__ void pack_w_bf16_kernel(const float* __restrict__ W, unsigned int* __restrict__ Wp,
                                   int K, int N) {
    int i = blockIdx.x * blockDim.x + threadIdx.x;   // over N*(K/2)
    int kh = K >> 1;
    if (i >= N * kh) return;
    int n = i / kh;
    int q = i - n * kh;
    union { __bf16 b[2]; unsigned int u; } cv;
    cv.b[0] = (__bf16)W[(size_t)(2 * q + 0) * N + n];
    cv.b[1] = (__bf16)W[(size_t)(2 * q + 1) * N + n];
    Wp[i] = cv.u;
}

// ---------------------------------------------------------------------------
// softmax over K=3 taps for H=16 heads
// ---------------------------------------------------------------------------
__global__ void softmax_convw_kernel(const float* __restrict__ w, float* __restrict__ out) {
    int h = threadIdx.x;
    if (h < 16) {
        float a = w[h * 3 + 0], b = w[h * 3 + 1], c = w[h * 3 + 2];
        float m = fmaxf(a, fmaxf(b, c));
        float ea = __expf(a - m), eb = __expf(b - m), ec = __expf(c - m);
        float s = ea + eb + ec;
        out[h * 3 + 0] = ea / s;
        out[h * 3 + 1] = eb / s;
        out[h * 3 + 2] = ec / s;
    }
}

// ---------------------------------------------------------------------------
// row LayerNorm (torch-style: unbiased std, eps added to std), bf16 output
// ---------------------------------------------------------------------------
__global__ __launch_bounds__(256)
void norm_rows_kernel(const float* __restrict__ x, const float* __restrict__ alpha,
                      const float* __restrict__ bias, __bf16* __restrict__ y, int ncols) {
    const int tid = threadIdx.x;
    const int row = blockIdx.x;
    const float* xr = x + (size_t)row * ncols;

    float s = 0.f, q = 0.f;
    for (int c = tid; c < ncols; c += 256) {
        float v = xr[c];
        s += v; q += v * v;
    }
    __shared__ float sh_s[256], sh_q[256];
    sh_s[tid] = s; sh_q[tid] = q;
    __syncthreads();
    for (int o = 128; o > 0; o >>= 1) {
        if (tid < o) { sh_s[tid] += sh_s[tid + o]; sh_q[tid] += sh_q[tid + o]; }
        __syncthreads();
    }
    float mu  = sh_s[0] / (float)ncols;
    float var = (sh_q[0] - (float)ncols * mu * mu) / (float)(ncols - 1);
    float sd  = sqrtf(var > 0.f ? var : 0.f);
    float inv = 1.f / (sd + 1e-6f);
    for (int c = tid; c < ncols; c += 256) {
        float v = alpha[c] * (xr[c] - mu) * inv + bias[c];
        y[(size_t)row * ncols + c] = (__bf16)v;
    }
}

// ---------------------------------------------------------------------------
// WMMA bf16 GEMM, double-buffered LDS software pipeline.
// A: [M,K] bf16 row-major.  Bp: packed uint[N][K/2].
// Epilogues:  Cf != nullptr : fp32 out = acc + bias + res (residual add)
//             Cf == nullptr : bf16 out = act(acc + bias)  (relu optional)
// Block 256 threads = 8 waves, tile 128x128, K-step 32; wave grid 2x4,
// each wave computes 64(M) x 32(N) = 4x2 WMMA accumulators.
// Pipeline: iter i -> issue global loads for tile i+1, WMMA tile i from
// LDS[buf], store staged regs to LDS[buf^1], single barrier.
// ---------------------------------------------------------------------------
__global__ __launch_bounds__(256)
void wmma_gemm_bf16_kernel(const __bf16* __restrict__ A, const unsigned int* __restrict__ Bp,
                           const float* __restrict__ bias, const float* __restrict__ res,
                           __bf16* __restrict__ Cb, float* __restrict__ Cf,
                           int M, int N, int K, int relu) {
    __shared__ __bf16       As[2][128 * 32];        // A tiles, row-major
    __shared__ unsigned int Bs[2][128 * BQ_STRIDE]; // B tiles, packed pairs per column

    const int tid  = threadIdx.x;
    const int lane = tid & 31;
    const int wv   = tid >> 5;
    const int wrow = wv >> 2;   // 0..1  (64-row slab)
    const int wcol = wv & 3;    // 0..3  (32-col slab)
    const int bm   = blockIdx.y * 128;
    const int bn   = blockIdx.x * 128;
    const int Kq   = K >> 1;    // column stride of Bp in uints

    // per-thread staging coordinates (2 chunks each for A and B)
    const int ch0 = tid * 2, ch1 = tid * 2 + 1;
    const int ra0 = ch0 >> 2, ca0 = ch0 & 3;   // A: 128 rows x 4 uint4/row
    const int ra1 = ch1 >> 2, ca1 = ch1 & 3;
    const int cb0 = ch0 >> 2, pb0 = ch0 & 3;   // B: 128 cols x 4 uint4/col
    const int cb1 = ch1 >> 2, pb1 = ch1 & 3;

    const __bf16* gA0 = A + (size_t)(bm + ra0) * K + ca0 * 8;
    const __bf16* gA1 = A + (size_t)(bm + ra1) * K + ca1 * 8;
    const unsigned int* gB0 = Bp + (size_t)(bn + cb0) * Kq + pb0 * 4;
    const unsigned int* gB1 = Bp + (size_t)(bn + cb1) * Kq + pb1 * 4;

    v8f acc[4][2];
#pragma unroll
    for (int mt = 0; mt < 4; ++mt)
#pragma unroll
        for (int nt = 0; nt < 2; ++nt)
            acc[mt][nt] = (v8f){0.f, 0.f, 0.f, 0.f, 0.f, 0.f, 0.f, 0.f};

    // ---- prologue: stage tile 0 into buffer 0 ----
    {
        uint4 va0 = *(const uint4*)(gA0);
        uint4 va1 = *(const uint4*)(gA1);
        uint4 vb0 = *(const uint4*)(gB0);
        uint4 vb1 = *(const uint4*)(gB1);
        *(uint4*)(As[0] + ra0 * 32 + ca0 * 8) = va0;
        *(uint4*)(As[0] + ra1 * 32 + ca1 * 8) = va1;
        *(uint4*)(Bs[0] + cb0 * BQ_STRIDE + pb0 * 4) = vb0;
        *(uint4*)(Bs[0] + cb1 * BQ_STRIDE + pb1 * 4) = vb1;
    }
    __syncthreads();

    const int kh = (lane < 16) ? 0 : 8;   // A uint col offset per lane half
    const int qh = (lane < 16) ? 0 : 8;   // B q offset per lane half

    for (int k0 = 0; k0 < K; k0 += 32) {
        const int buf = (k0 >> 5) & 1;
        const bool has_next = (k0 + 32 < K);

        // ---- issue next tile's global loads first (latency hidden by WMMAs) ----
        uint4 va0, va1, vb0, vb1;
        if (has_next) {
            const int kn = k0 + 32, qn = kn >> 1;
            va0 = *(const uint4*)(gA0 + kn);
            va1 = *(const uint4*)(gA1 + kn);
            vb0 = *(const uint4*)(gB0 + qn);
            vb1 = *(const uint4*)(gB1 + qn);
            if (k0 + 64 < K) {   // prefetch tile i+2 (global_prefetch_b8)
                __builtin_prefetch(gA0 + kn + 32, 0, 3);
                __builtin_prefetch(gB0 + qn + 16, 0, 3);
            }
        }

        // ---- fragments: all contiguous 16B LDS reads ----
        v16bf afr[4], bfr[2];
        const unsigned int* Asu = (const unsigned int*)As[buf];
#pragma unroll
        for (int mt = 0; mt < 4; ++mt) {
            int m = wrow * 64 + mt * 16 + (lane & 15);
            unsigned int* au = (unsigned int*)&afr[mt];
#pragma unroll
            for (int p = 0; p < 4; ++p) au[p]     = Asu[m * 16 + kh + p];
#pragma unroll
            for (int p = 0; p < 4; ++p) au[p + 4] = Asu[m * 16 + kh + 4 + p];
        }
#pragma unroll
        for (int nt = 0; nt < 2; ++nt) {
            int n = wcol * 32 + nt * 16 + (lane & 15);
            const unsigned int* src = Bs[buf] + n * BQ_STRIDE + qh;
            unsigned int* bu = (unsigned int*)&bfr[nt];
#pragma unroll
            for (int p = 0; p < 8; ++p) bu[p] = src[p];
        }

        // ---- 8 WMMAs: D = A*B + C, f32 accumulate ----
#pragma unroll
        for (int mt = 0; mt < 4; ++mt)
#pragma unroll
            for (int nt = 0; nt < 2; ++nt)
                acc[mt][nt] = __builtin_amdgcn_wmma_f32_16x16x32_bf16(
                    false, afr[mt], false, bfr[nt], (short)0, acc[mt][nt], false, false);

        // ---- store staged regs into the other buffer, one barrier per iter ----
        if (has_next) {
            const int nb = buf ^ 1;
            *(uint4*)(As[nb] + ra0 * 32 + ca0 * 8) = va0;
            *(uint4*)(As[nb] + ra1 * 32 + ca1 * 8) = va1;
            *(uint4*)(Bs[nb] + cb0 * BQ_STRIDE + pb0 * 4) = vb0;
            *(uint4*)(Bs[nb] + cb1 * BQ_STRIDE + pb1 * 4) = vb1;
        }
        __syncthreads();
    }

    // ---- epilogue. C/D layout: VGPR r -> M=r (+8 for lanes 16-31), N=lane%16 ----
#pragma unroll
    for (int mt = 0; mt < 4; ++mt) {
        int rowb = bm + wrow * 64 + mt * 16 + ((lane >= 16) ? 8 : 0);
#pragma unroll
        for (int nt = 0; nt < 2; ++nt) {
            int n = bn + wcol * 32 + nt * 16 + (lane & 15);
            float bv = bias[n];
#pragma unroll
            for (int r = 0; r < 8; ++r) {
                size_t idx = (size_t)(rowb + r) * N + n;
                float v = acc[mt][nt][r] + bv;
                if (Cf) {
                    Cf[idx] = v + res[idx];
                } else {
                    if (relu) v = v > 0.f ? v : 0.f;
                    Cb[idx] = (__bf16)v;
                }
            }
        }
    }
}

// ---------------------------------------------------------------------------
// GLU + mask: h = a * sigmoid(b), zero where mask==0. g: [M, 2048] bf16
// ---------------------------------------------------------------------------
__global__ void glu_mask_kernel(const __bf16* __restrict__ g, const int* __restrict__ mask,
                                __bf16* __restrict__ h, int total) {
    int i = blockIdx.x * blockDim.x + threadIdx.x;
    if (i >= total) return;
    int m = i >> 10;           // row (b*S + s)
    int c = i & 1023;
    float a = (float)g[(size_t)m * 2048 + c];
    float b = (float)g[(size_t)m * 2048 + 1024 + c];
    float v = a * (1.f / (1.f + __expf(-b)));
    h[i] = (__bf16)((mask[m] != 0) ? v : 0.f);
}

// ---------------------------------------------------------------------------
// depthwise lightweight conv, K=3, pad (1,1), per-head (64 ch/head) weights
// ---------------------------------------------------------------------------
__global__ void lightconv_kernel(const __bf16* __restrict__ h, const float* __restrict__ w,
                                 __bf16* __restrict__ y) {
    size_t i = (size_t)blockIdx.x * blockDim.x + threadIdx.x; // < 16*1024*1024
    int c = (int)(i & 1023);
    size_t bs = i >> 10;
    int s = (int)(bs & 1023);
    int head = c >> 6;
    const float* wk = w + head * 3;
    float acc = wk[1] * (float)h[i];
    if (s > 0)    acc += wk[0] * (float)h[i - 1024];
    if (s < 1023) acc += wk[2] * (float)h[i + 1024];
    y[i] = (__bf16)acc;
}

// ---------------------------------------------------------------------------
// host-side launch
// ---------------------------------------------------------------------------
extern "C" void kernel_launch(void* const* d_in, const int* in_sizes, int n_in,
                              void* d_out, int out_size, void* d_ws, size_t ws_size,
                              hipStream_t stream) {
    (void)in_sizes; (void)n_in; (void)out_size; (void)ws_size;
    const int BSr = 16384;      // B*S rows
    const int D   = 1024;
    const int DFF = 4096;

    const float* x      = (const float*)d_in[0];
    const int*   mask   = (const int*)  d_in[1];
    const float* alpha1 = (const float*)d_in[2];
    const float* bias1  = (const float*)d_in[3];
    const float* alpha2 = (const float*)d_in[4];
    const float* bias2  = (const float*)d_in[5];
    const float* W_l1   = (const float*)d_in[6];
    const float* b_l1   = (const float*)d_in[7];
    const float* conv_w = (const float*)d_in[8];
    const float* W_l2   = (const float*)d_in[9];
    const float* b_l2   = (const float*)d_in[10];
    const float* W_ff1  = (const float*)d_in[11];
    const float* b_ff1  = (const float*)d_in[12];
    const float* W_ff2  = (const float*)d_in[13];
    const float* b_ff2  = (const float*)d_in[14];
    float* out = (float*)d_out;

    // workspace layout (bytes)
    char* ws = (char*)d_ws;
    __bf16*       p_xn  = (__bf16*)      (ws + 0);                    // 33,554,432  (norm out; later conv out; later norm2 out)
    unsigned int* p_wl1 = (unsigned int*)(ws + 33554432);             //  4,194,304  packed
    unsigned int* p_wl2 = (unsigned int*)(ws + 37748736);             //  2,097,152  packed
    unsigned int* p_wf1 = (unsigned int*)(ws + 39845888);             //  8,388,608  packed
    unsigned int* p_wf2 = (unsigned int*)(ws + 48234496);             //  8,388,608  packed
    float*        p_cw  = (float*)       (ws + 56623104);             //        256
    __bf16*       p_big = (__bf16*)      (ws + 56623360);             // 134,217,728 (gemm1 out / ff1 out)
    __bf16*       p_h   = (__bf16*)      (ws + 56623360 + 67108864);  // 33,554,432  (GLU out, in p_big's tail)

    // 1) weight conversions (transpose + pair-pack into WMMA B layout)
    pack_w_bf16_kernel<<<(2 * D * (D / 2) + 255) / 256, 256, 0, stream>>>(W_l1, p_wl1, D, 2 * D);
    pack_w_bf16_kernel<<<(D * (D / 2) + 255) / 256, 256, 0, stream>>>(W_l2, p_wl2, D, D);
    pack_w_bf16_kernel<<<(DFF * (D / 2) + 255) / 256, 256, 0, stream>>>(W_ff1, p_wf1, D, DFF);
    pack_w_bf16_kernel<<<(D * (DFF / 2) + 255) / 256, 256, 0, stream>>>(W_ff2, p_wf2, DFF, D);

    // 2) conv-weight softmax
    softmax_convw_kernel<<<1, 64, 0, stream>>>(conv_w, p_cw);

    // 3) norm1 -> bf16
    norm_rows_kernel<<<BSr, 256, 0, stream>>>(x, alpha1, bias1, p_xn, D);

    // 4) GEMM1: xn @ W_l1 + b_l1  -> bf16 [BSr, 2048]
    wmma_gemm_bf16_kernel<<<dim3(2 * D / 128, BSr / 128), 256, 0, stream>>>(
        p_xn, p_wl1, b_l1, nullptr, p_big, nullptr, BSr, 2 * D, D, 0);

    // 5) GLU + mask -> bf16 [BSr, 1024]
    glu_mask_kernel<<<(BSr * D + 255) / 256, 256, 0, stream>>>(p_big, mask, p_h, BSr * D);

    // 6) lightweight conv -> bf16 (reuse p_xn)
    lightconv_kernel<<<(BSr * D) / 256, 256, 0, stream>>>(p_h, p_cw, p_xn);

    // 7) GEMM2: conv @ W_l2 + b_l2 + x -> fp32 d_out (x1)
    wmma_gemm_bf16_kernel<<<dim3(D / 128, BSr / 128), 256, 0, stream>>>(
        p_xn, p_wl2, b_l2, x, nullptr, out, BSr, D, D, 0);

    // 8) norm2(x1) -> bf16 (reuse p_xn)
    norm_rows_kernel<<<BSr, 256, 0, stream>>>(out, alpha2, bias2, p_xn, D);

    // 9) GEMM3: xn2 @ W_ff1 + b_ff1, ReLU -> bf16 [BSr, 4096] (reuse p_big)
    wmma_gemm_bf16_kernel<<<dim3(DFF / 128, BSr / 128), 256, 0, stream>>>(
        p_xn, p_wf1, b_ff1, nullptr, p_big, nullptr, BSr, DFF, D, 1);

    // 10) GEMM4: relu @ W_ff2 + b_ff2 + x1 -> fp32 d_out (in-place residual)
    wmma_gemm_bf16_kernel<<<dim3(D / 128, BSr / 128), 256, 0, stream>>>(
        p_big, p_wf2, b_ff2, out, nullptr, out, BSr, D, DFF, 0);
}
